// MultiDecoder_66975720014163
// MI455X (gfx1250) — matrix operand order repaired and checked by reading, not compiled, not run on hardware
//
#include <hip/hip_runtime.h>

// ---------------- types ----------------
typedef __bf16 bf16;
typedef __attribute__((ext_vector_type(16))) __bf16 v16bf;
typedef __attribute__((ext_vector_type(8)))  __bf16 bf16x8;
typedef __attribute__((ext_vector_type(4)))  __bf16 bf16x4;
typedef __attribute__((ext_vector_type(8)))  float  v8f;

// ---------------- problem constants ----------------
enum { B_ = 16, T_ = 256, S_ = 256, NA_ = 300, D_ = 512, H_ = 8, DK_ = 64,
       DFF_ = 2048, L_ = 6, V_ = 32000 };

// ---------------- GEMM tiling ----------------
#define BM 128
#define BN 64
#define BK 32
#define LDS_K 40   // BK + 8 pad (80B row stride -> 16B aligned sub-offsets)

struct GemmP {
  const float* A; const float* B; float* C;
  const float* biasRow; const float* biasCol;
  int M, N, K;
  int lda, ldb, ldc;
  int bdiv;                     // batch z -> (z/bdiv, z%bdiv)
  long long sA1, sA2, sB1, sB2, sC1, sC2;
  float alpha;
  int transB, accum, relu;
};

// branchless guarded load: clamped address + select
__device__ __forceinline__ float ldg_guard(const float* __restrict__ p,
                                           long long idx, bool ok) {
  const float v = p[ok ? idx : 0];
  return ok ? v : 0.f;
}

// C[M,N] = alpha * A[M,K] @ B[K,N] (+C) (+biasRow[m]) (+biasCol[n]) (relu)
// fp32 in/out, bf16 WMMA compute. 128 threads = 4 waves; wave tile 64x32
// (4x2 WMMA 16x16x32 tiles), block tile 128x64.
__global__ __launch_bounds__(128) void gemm_bf16_wmma(GemmP p) {
  const int z  = blockIdx.z;
  const int z1 = z / p.bdiv;
  const int z2 = z % p.bdiv;
  const float* A  = p.A + z1 * p.sA1 + z2 * p.sA2;
  const float* Bp = p.B + z1 * p.sB1 + z2 * p.sB2;
  float*       C  = p.C + z1 * p.sC1 + z2 * p.sC2;

  const int m0 = blockIdx.y * BM;
  const int n0 = blockIdx.x * BN;

  __shared__ bf16 As[BM][LDS_K];   // A tile, row-major [m][k]
  __shared__ bf16 BsT[BN][LDS_K];  // B tile transposed [n][k]

  const int tid  = threadIdx.x;
  const int wave = tid >> 5;
  const int lane = tid & 31;
  const int wm   = (wave >> 1) * 64;   // 0 / 64
  const int wn   = (wave & 1) * 32;    // 0 / 32
  const int lrow = lane & 15;
  const int lhi  = lane >> 4;

  const v8f zero8 = {0.f,0.f,0.f,0.f,0.f,0.f,0.f,0.f};
  v8f acc[4][2];
#pragma unroll
  for (int i = 0; i < 4; ++i) { acc[i][0] = zero8; acc[i][1] = zero8; }

  for (int k0 = 0; k0 < p.K; k0 += BK) {
    // prefetch next K tile into cache while this one is consumed
    if (k0 + BK < p.K) {
      __builtin_prefetch(A + (long long)(m0 + (tid & 127)) * p.lda + (k0 + BK), 0, 3);
      if (p.transB)
        __builtin_prefetch(Bp + (long long)(n0 + (tid & 63)) * p.ldb + (k0 + BK), 0, 3);
      else
        __builtin_prefetch(Bp + (long long)(k0 + BK + (tid & 31)) * p.ldb + n0, 0, 3);
    }

    const bool fullA = (m0 + BM <= p.M) && (k0 + BK <= p.K);
    const bool fullB = (n0 + BN <= p.N) && (k0 + BK <= p.K);

    // ---- A tile: 128 rows x 32 k, packed 8B LDS stores ----
#pragma unroll
    for (int it = 0; it < 8; ++it) {
      const int qi = tid + 128 * it;          // 0..1023 float4-groups
      const int r  = qi >> 3;
      const int c4 = (qi & 7) << 2;
      bf16x4 pk;
      if (fullA) {
        const float4 f = *(const float4*)(A + (long long)(m0 + r) * p.lda + (k0 + c4));
        pk[0] = (bf16)f.x; pk[1] = (bf16)f.y; pk[2] = (bf16)f.z; pk[3] = (bf16)f.w;
      } else {
        const int m = m0 + r;
        const bool mok = m < p.M;
#pragma unroll
        for (int j = 0; j < 4; ++j) {
          const int k = k0 + c4 + j;
          pk[j] = (bf16)ldg_guard(A, (long long)m * p.lda + k, mok && (k < p.K));
        }
      }
      *(bf16x4*)&As[r][c4] = pk;
    }

    // ---- B tile -> BsT[n][k], packed 8B LDS stores ----
    if (p.transB) {   // B(k,n) = Bp[n*ldb + k] : contiguous along k
#pragma unroll
      for (int it = 0; it < 4; ++it) {
        const int qi = tid + 128 * it;        // 0..511
        const int n  = qi >> 3;
        const int c4 = (qi & 7) << 2;
        bf16x4 pk;
        if (fullB) {
          const float4 f = *(const float4*)(Bp + (long long)(n0 + n) * p.ldb + (k0 + c4));
          pk[0] = (bf16)f.x; pk[1] = (bf16)f.y; pk[2] = (bf16)f.z; pk[3] = (bf16)f.w;
        } else {
          const int nn = n0 + n;
          const bool nok = nn < p.N;
#pragma unroll
          for (int j = 0; j < 4; ++j) {
            const int k = k0 + c4 + j;
            pk[j] = (bf16)ldg_guard(Bp, (long long)nn * p.ldb + k, nok && (k < p.K));
          }
        }
        *(bf16x4*)&BsT[n][c4] = pk;
      }
    } else {          // B(k,n) = Bp[k*ldb + n] : lane-coalesced along n
#pragma unroll
      for (int it = 0; it < 4; ++it) {
        const int qi = tid + 128 * it;        // 0..511
        const int n  = qi & 63;
        const int k4 = (qi >> 6) << 2;        // 0,4,...,28
        const int nn = n0 + n;
        const bool nok = nn < p.N;
        bf16x4 pk;
#pragma unroll
        for (int j = 0; j < 4; ++j) {
          const int k = k0 + k4 + j;
          pk[j] = (bf16)ldg_guard(Bp, (long long)k * p.ldb + nn, nok && (k < p.K));
        }
        *(bf16x4*)&BsT[n][k4] = pk;
      }
    }
    __syncthreads();

    // ---- fragments per the documented 16x16x32 bf16 striping ----
    // A lane(l<16): row=l,  K = [lhi*8 .. +7] then [lhi*8+16 .. +23]
    v16bf afrag[4], bfrag[2];
#pragma unroll
    for (int i = 0; i < 4; ++i) {
      const bf16x8 lo = *(const bf16x8*)&As[wm + i * 16 + lrow][lhi * 8];
      const bf16x8 hi = *(const bf16x8*)&As[wm + i * 16 + lrow][lhi * 8 + 16];
#pragma unroll
      for (int e = 0; e < 8; ++e) { afrag[i][e] = lo[e]; afrag[i][e + 8] = hi[e]; }
    }
    // B lane: col = lrow, K = [lhi*8 .. +7] then [lhi*8+16 .. +23]
#pragma unroll
    for (int j = 0; j < 2; ++j) {
      const bf16x8 lo = *(const bf16x8*)&BsT[wn + j * 16 + lrow][lhi * 8];
      const bf16x8 hi = *(const bf16x8*)&BsT[wn + j * 16 + lrow][lhi * 8 + 16];
#pragma unroll
      for (int e = 0; e < 8; ++e) { bfrag[j][e] = lo[e]; bfrag[j][e + 8] = hi[e]; }
    }
#pragma unroll
    for (int i = 0; i < 4; ++i)
#pragma unroll
      for (int j = 0; j < 2; ++j)
        acc[i][j] = __builtin_amdgcn_wmma_f32_16x16x32_bf16(
            false, afrag[i], false, bfrag[j], (short)0, acc[i][j], false, false);
    __syncthreads();
  }

  // ---- epilogue: C/D layout: VGPR r -> M = r + lhi*8, N = lrow ----
#pragma unroll
  for (int i = 0; i < 4; ++i) {
#pragma unroll
    for (int j = 0; j < 2; ++j) {
      const int gn = n0 + wn + j * 16 + lrow;
      if (gn >= p.N) continue;
      const int gmb = m0 + wm + i * 16 + lhi * 8;
#pragma unroll
      for (int r = 0; r < 8; ++r) {
        const int gm = gmb + r;
        if (gm >= p.M) continue;
        float val = acc[i][j][r] * p.alpha;
        const long long ci = (long long)gm * p.ldc + gn;
        if (p.accum)   val += C[ci];
        if (p.biasCol) val += p.biasCol[gn];
        if (p.biasRow) val += p.biasRow[gm];
        if (p.relu)    val = fmaxf(val, 0.f);
        C[ci] = val;
      }
    }
  }
}

// ---------------- masked softmax over 256-wide rows (in place) ----------------
// mode 0: causal + pad(dec); mode 1: pad(enc); mode 2: none
__global__ __launch_bounds__(256) void softmax_mask(float* __restrict__ data, int rows,
                                                    int mode, const int* __restrict__ ids,
                                                    int perB, int Tdim) {
  const int wave = threadIdx.x >> 5;
  const int lane = threadIdx.x & 31;
  const long long row = (long long)blockIdx.x * 8 + wave;
  if (row >= rows) return;
  float* d = data + row * 256;
  int b = 0, qi = 0;
  if (mode == 0) { b = (int)(row / perB); qi = (int)(row % Tdim); }
  else if (mode == 1) { b = (int)(row / perB); }
  float v[8];
#pragma unroll
  for (int c = 0; c < 8; ++c) {
    const int j = lane + 32 * c;
    float val = d[j];
    bool msk = false;
    if (mode == 0)      msk = (j > qi) || (ids[b * Tdim + j] == 0);
    else if (mode == 1) msk = (ids[b * Tdim + j] == 0);
    v[c] = msk ? -1e9f : val;
  }
  float m = v[0];
#pragma unroll
  for (int c = 1; c < 8; ++c) m = fmaxf(m, v[c]);
#pragma unroll
  for (int off = 16; off > 0; off >>= 1) m = fmaxf(m, __shfl_xor(m, off, 32));
  float s = 0.f;
#pragma unroll
  for (int c = 0; c < 8; ++c) { v[c] = __expf(v[c] - m); s += v[c]; }
#pragma unroll
  for (int off = 16; off > 0; off >>= 1) s += __shfl_xor(s, off, 32);
  const float inv = 1.f / s;
#pragma unroll
  for (int c = 0; c < 8; ++c) d[lane + 32 * c] = v[c] * inv;
}

// ---------------- residual LayerNorm over D=512, wave per row ----------------
__global__ __launch_bounds__(256) void ln_res(const float* __restrict__ resid,
                                              const float* __restrict__ inp,
                                              float* __restrict__ outp,
                                              const float* __restrict__ g,
                                              const float* __restrict__ b, int rows) {
  const int wave = threadIdx.x >> 5;
  const int lane = threadIdx.x & 31;
  const long long row = (long long)blockIdx.x * 8 + wave;
  if (row >= rows) return;
  const float* r0 = resid + row * (long long)D_;
  const float* i0 = inp + row * (long long)D_;
  float v[16];
  float s = 0.f, sq = 0.f;
#pragma unroll
  for (int qd = 0; qd < 4; ++qd) {
    const int col = lane * 4 + 128 * qd;
    float4 a = *(const float4*)(r0 + col);
    const float4 c = *(const float4*)(i0 + col);
    a.x += c.x; a.y += c.y; a.z += c.z; a.w += c.w;
    v[qd * 4 + 0] = a.x; v[qd * 4 + 1] = a.y; v[qd * 4 + 2] = a.z; v[qd * 4 + 3] = a.w;
    s += a.x + a.y + a.z + a.w;
    sq += a.x * a.x + a.y * a.y + a.z * a.z + a.w * a.w;
  }
#pragma unroll
  for (int off = 16; off > 0; off >>= 1) {
    s  += __shfl_xor(s, off, 32);
    sq += __shfl_xor(sq, off, 32);
  }
  const float mu   = s * (1.f / (float)D_);
  const float var  = sq * (1.f / (float)D_) - mu * mu;   // biased, as torch LN
  const float rstd = rsqrtf(var + 1e-5f);
  float* o0 = outp + row * (long long)D_;
#pragma unroll
  for (int qd = 0; qd < 4; ++qd) {
    const int col = lane * 4 + 128 * qd;
    float4 o;
    float t0 = (v[qd * 4 + 0] - mu) * rstd;
    float t1 = (v[qd * 4 + 1] - mu) * rstd;
    float t2 = (v[qd * 4 + 2] - mu) * rstd;
    float t3 = (v[qd * 4 + 3] - mu) * rstd;
    if (g) {
      t0 = t0 * g[col + 0] + b[col + 0];
      t1 = t1 * g[col + 1] + b[col + 1];
      t2 = t2 * g[col + 2] + b[col + 2];
      t3 = t3 * g[col + 3] + b[col + 3];
    }
    o.x = t0; o.y = t1; o.z = t2; o.w = t3;
    *(float4*)(o0 + col) = o;
  }
}

// ---------------- embedding + sinusoidal positional encoding ----------------
__global__ __launch_bounds__(256) void embed_pe(const int* __restrict__ dec,
                                                const float* __restrict__ emb,
                                                float* __restrict__ x, int total) {
  const int idx = blockIdx.x * blockDim.x + threadIdx.x;
  if (idx >= total) return;
  const int d   = idx & (D_ - 1);
  const int row = idx >> 9;              // D_=512
  const int t   = row & (T_ - 1);
  const int tok = dec[row];
  const int pr  = d >> 1;
  const float div = __expf(-__logf(10000.f) * (float)(2 * pr) * (1.f / (float)D_));
  const float ang = (float)t * div;
  const float pe  = (d & 1) ? __cosf(ang) : __sinf(ang);
  x[idx] = emb[(long long)tok * D_ + d] + pe;
}

// ---------------- host helpers ----------------
static void gemm(hipStream_t s, const float* A, const float* B, float* C,
                 int M, int N, int K, int lda, int ldb, int ldc,
                 int batches, int bdiv,
                 long long sA1, long long sA2, long long sB1, long long sB2,
                 long long sC1, long long sC2,
                 float alpha, int transB, int accum, int relu,
                 const float* biasRow, const float* biasCol) {
  GemmP p{A, B, C, biasRow, biasCol, M, N, K, lda, ldb, ldc, bdiv,
          sA1, sA2, sB1, sB2, sC1, sC2, alpha, transB, accum, relu};
  dim3 grid((N + BN - 1) / BN, (M + BM - 1) / BM, batches);
  gemm_bf16_wmma<<<grid, dim3(128, 1, 1), 0, s>>>(p);
}

static void softmax(hipStream_t s, float* data, int rows, int mode,
                    const int* ids, int perB, int Tdim) {
  softmax_mask<<<dim3((rows + 7) / 8), dim3(256), 0, s>>>(data, rows, mode, ids, perB, Tdim);
}

static void layernorm(hipStream_t s, const float* resid, const float* inp, float* outp,
                      const float* g, const float* b, int rows) {
  ln_res<<<dim3((rows + 7) / 8), dim3(256), 0, s>>>(resid, inp, outp, g, b, rows);
}

extern "C" void kernel_launch(void* const* d_in, const int* in_sizes, int n_in,
                              void* d_out, int out_size, void* d_ws, size_t ws_size,
                              hipStream_t stream) {
  (void)in_sizes; (void)n_in; (void)out_size; (void)ws_size;
  typedef long long ll;
  const int*   dec     = (const int*)d_in[0];
  const int*   enc_ids = (const int*)d_in[1];
  const float* enc_out = (const float*)d_in[2];
  const float* ast_out = (const float*)d_in[3];
  const float* src_emb = (const float*)d_in[4];
  const float* ast_emb = (const float*)d_in[5];
  /* d_in[6] match_rela: unused by reference */
  const float* emb     = (const float*)d_in[7];
  const float* attn_W  = (const float*)d_in[8];
  const float* lng     = (const float*)d_in[9];
  const float* lnb     = (const float*)d_in[10];
  const float* fW1     = (const float*)d_in[11];
  const float* fW2     = (const float*)d_in[12];
  const float* conv_w  = (const float*)d_in[13];
  const float* conv_b  = (const float*)d_in[14];
  const float* mW      = (const float*)d_in[15];
  const float* mb      = (const float*)d_in[16];

  // ---- workspace carve-up ----
  float* w = (float*)d_ws;
  size_t o = 0;
  float* x    = w + o; o += (size_t)B_ * T_ * D_;
  float* q    = w + o; o += (size_t)B_ * T_ * D_;
  float* kbuf = w + o; o += (size_t)B_ * S_ * D_;
  float* vbuf = w + o; o += (size_t)B_ * S_ * D_;
  float* ctx  = w + o; o += (size_t)B_ * T_ * D_;
  float* proj = w + o; o += (size_t)B_ * T_ * D_;
  float* ffnh = w + o; o += (size_t)B_ * T_ * DFF_;
  float* mout = w + o; o += (size_t)B_ * S_ * D_;
  float* ast1 = w + o; o += (size_t)B_ * S_ * D_;
  float* aste = w + o; o += (size_t)B_ * S_ * D_;
  float* mmsc = w + o; o += (size_t)B_ * S_ * S_;

  // ---- output carve-up ----
  float* out   = (float*)d_out;
  const ll CHUNK = (ll)B_ * H_ * T_ * T_;      // per-layer attn slice (Tk=256)
  float* selfA = out + (ll)B_ * T_ * D_;
  float* encA  = selfA + (ll)L_ * CHUNK;
  float* astA  = encA + (ll)L_ * CHUNK;

  // ============ multi-model fusion block ============
  // ast1 = conv_w @ ast_outputs + conv_b  (batched over B; A shared)
  gemm(stream, conv_w, ast_out, ast1, S_, D_, NA_, NA_, D_, D_,
       B_, 1, 0, 0, (ll)NA_ * D_, 0, (ll)S_ * D_, 0,
       1.f, 0, 0, 0, conv_b, nullptr);
  // ast_e1 = conv_w @ ast_embed + conv_b
  gemm(stream, conv_w, ast_emb, aste, S_, D_, NA_, NA_, D_, D_,
       B_, 1, 0, 0, (ll)NA_ * D_, 0, (ll)S_ * D_, 0,
       1.f, 0, 0, 0, conv_b, nullptr);
  // mm_scores = ast1 @ enc_out^T / sqrt(DK)
  gemm(stream, ast1, enc_out, mmsc, S_, S_, D_, D_, D_, S_,
       B_, 1, (ll)S_ * D_, 0, (ll)S_ * D_, 0, (ll)S_ * S_, 0,
       0.125f, 1, 0, 0, nullptr, nullptr);
  softmax(stream, mmsc, B_ * S_, 2, nullptr, 1, 256);
  // multi_out = src_embed @ W_top + b
  gemm(stream, src_emb, mW, mout, B_ * S_, D_, D_, D_, D_, D_,
       1, 1, 0, 0, 0, 0, 0, 0, 1.f, 0, 0, 0, nullptr, mb);
  // multi_out += ast_e1 @ W_bot
  gemm(stream, aste, mW + (ll)D_ * D_, mout, B_ * S_, D_, D_, D_, D_, D_,
       1, 1, 0, 0, 0, 0, 0, 0, 1.f, 0, 1, 0, nullptr, nullptr);
  // multi_out += softmax(mm_scores) @ enc_out
  gemm(stream, mmsc, enc_out, mout, S_, D_, S_, S_, D_, D_,
       B_, 1, (ll)S_ * S_, 0, (ll)S_ * D_, 0, (ll)S_ * D_, 0,
       1.f, 0, 1, 0, nullptr, nullptr);

  // ============ decoder input: embedding + PE ============
  {
    const int total = B_ * T_ * D_;
    embed_pe<<<dim3((total + 255) / 256), dim3(256), 0, stream>>>(dec, emb, x, total);
  }

  // ============ decoder layers ============
  auto mha = [&](const float* kvsrc, int Tk, const float* W, float* attnOut,
                 int mode, const int* ids, const float* g, const float* bb) {
    // q/k/v projections (M folds batch since x / kv are [B*T, D] contiguous)
    gemm(stream, x,     W + 0 * (ll)D_ * D_, q,    B_ * T_,  D_, D_, D_, D_, D_,
         1, 1, 0,0,0,0,0,0, 1.f, 0, 0, 0, nullptr, nullptr);
    gemm(stream, kvsrc, W + 1 * (ll)D_ * D_, kbuf, B_ * Tk,  D_, D_, D_, D_, D_,
         1, 1, 0,0,0,0,0,0, 1.f, 0, 0, 0, nullptr, nullptr);
    gemm(stream, kvsrc, W + 2 * (ll)D_ * D_, vbuf, B_ * Tk,  D_, D_, D_, D_, D_,
         1, 1, 0,0,0,0,0,0, 1.f, 0, 0, 0, nullptr, nullptr);
    // scores = q @ k^T / 8, batched over (b,h), straight into d_out
    gemm(stream, q, kbuf, attnOut, T_, Tk, DK_, D_, D_, Tk,
         B_ * H_, H_, (ll)T_ * D_, DK_, (ll)Tk * D_, DK_,
         (ll)H_ * T_ * Tk, (ll)T_ * Tk,
         0.125f, 1, 0, 0, nullptr, nullptr);
    softmax(stream, attnOut, B_ * H_ * T_, mode, ids, H_ * T_, 256);
    // ctx = attn @ v
    gemm(stream, attnOut, vbuf, ctx, T_, DK_, Tk, Tk, D_, D_,
         B_ * H_, H_, (ll)H_ * T_ * Tk, (ll)T_ * Tk, (ll)Tk * D_, DK_,
         (ll)T_ * D_, DK_, 1.f, 0, 0, 0, nullptr, nullptr);
    // out projection + residual LN (x updated in place)
    gemm(stream, ctx, W + 3 * (ll)D_ * D_, proj, B_ * T_, D_, D_, D_, D_, D_,
         1, 1, 0,0,0,0,0,0, 1.f, 0, 0, 0, nullptr, nullptr);
    layernorm(stream, x, proj, x, g, bb, B_ * T_);
  };

  for (int l = 0; l < L_; ++l) {
    const float* Wl  = attn_W + (ll)l * 3 * 4 * D_ * D_;
    const float* g0  = lng + (ll)(l * 3 + 0) * D_;
    const float* b0  = lnb + (ll)(l * 3 + 0) * D_;
    const float* g1  = lng + (ll)(l * 3 + 1) * D_;
    const float* b1  = lnb + (ll)(l * 3 + 1) * D_;
    const float* g2  = lng + (ll)(l * 3 + 2) * D_;
    const float* b2  = lnb + (ll)(l * 3 + 2) * D_;

    mha(x,       T_, Wl + 0 * 4 * (ll)D_ * D_, selfA + (ll)l * CHUNK, 0, dec,     g0, b0);
    mha(enc_out, S_, Wl + 1 * 4 * (ll)D_ * D_, encA  + (ll)l * CHUNK, 1, enc_ids, g1, b1);
    mha(mout,    S_, Wl + 2 * 4 * (ll)D_ * D_, astA  + (ll)l * CHUNK, 2, nullptr, g2, b2);

    // FFN: h = relu(x @ W1); proj = h @ W2; x = LN(x + proj) with default params
    gemm(stream, x, fW1 + (ll)l * D_ * DFF_, ffnh, B_ * T_, DFF_, D_, D_, DFF_, DFF_,
         1, 1, 0,0,0,0,0,0, 1.f, 0, 0, 1, nullptr, nullptr);
    gemm(stream, ffnh, fW2 + (ll)l * DFF_ * D_, proj, B_ * T_, D_, DFF_, DFF_, D_, D_,
         1, 1, 0,0,0,0,0,0, 1.f, 0, 0, 0, nullptr, nullptr);
    layernorm(stream, x, proj, x, nullptr, nullptr, B_ * T_);
  }

  // final hidden state -> d_out[0 : B*T*D]
  hipMemcpyAsync(out, x, (size_t)B_ * T_ * D_ * sizeof(float),
                 hipMemcpyDeviceToDevice, stream);
}
